// CombinedSurvLoss_89172110999808
// MI455X (gfx1250) — matrix operand order repaired and checked by reading, not compiled, not run on hardware
//
#include <hip/hip_runtime.h>
#include <hip/hip_bf16.h>
#include <math.h>

typedef __attribute__((ext_vector_type(16))) _Float16 v16h;
typedef __attribute__((ext_vector_type(8)))  float    v8f;

#define EPS   1e-7f
#define NBINS 4
#define PANEL 2048   // j-panel staged in LDS: 2048*(4+2+2)B = 16 KB

// ---------------------------------------------------------------------------
// Kernel 1: per-row survival quantities.
//   hazards = sigmoid(outputs); S = cumprod(1-h); risk = -sum(S)
//   w = exp(risk) split into f16 hi+lo for near-fp32 WMMA accumulation.
//   nll per row per the reference (alpha=0).
// ---------------------------------------------------------------------------
__global__ void surv_row_kernel(const float* __restrict__ outputs,
                                const int* __restrict__ y,
                                const int* __restrict__ c,
                                float* __restrict__ risk,
                                _Float16* __restrict__ whi,
                                _Float16* __restrict__ wlo,
                                float* __restrict__ nll,
                                int B) {
    int i = blockIdx.x * blockDim.x + threadIdx.x;
    if (i >= B) return;

    float h[NBINS];
    float S[NBINS + 1];
    S[0] = 1.0f;
#pragma unroll
    for (int k = 0; k < NBINS; ++k) {
        float x = outputs[i * NBINS + k];
        h[k] = 1.0f / (1.0f + __expf(-x));
        S[k + 1] = S[k] * (1.0f - h[k]);
    }
    float r = -(S[1] + S[2] + S[3] + S[4]);
    risk[i] = r;

    float w = __expf(r);                 // risk in (-4,0) -> w in (e^-4, 1): safe, no shift
    _Float16 wh = (_Float16)w;
    whi[i] = wh;
    wlo[i] = (_Float16)(w - (float)wh);  // residual for 2-term WMMA accumulation

    int   yi = y[i];
    float cf = (float)c[i];
    float s_prev = fmaxf(S[yi],     EPS);
    float h_this = fmaxf(h[yi],     EPS);
    float s_this = fmaxf(S[yi + 1], EPS);
    nll[i] = -cf * __logf(s_this)
             - (1.0f - cf) * (__logf(s_prev) + __logf(h_this));
}

// ---------------------------------------------------------------------------
// Kernel 2: pairwise masked exp-sum via WMMA.
//   pairSum[i] = sum_j [t_j > t_i] * exp(risk_j)
// Each wave (wave32) owns 16 i-rows. A = {0,1} mask (16x32 f16, exact),
// B = w broadcast across 16 columns. Two WMMAs per chunk (hi + lo halves of w)
// accumulate into one f32 C for fp32-grade accuracy.
// VGPR layouts per CDNA5 ISA 7.12.2 (wave32):
//   A 16x32 f16: lane L, M=L%16; lanes 0-15 hold K = {2v,2v+1} (v<4) and
//                {16+2(v-4),...} (v>=4); lanes 16-31 the same +8 / +24.
//   B 32x16 f16: same K order as A (lanes 0-15 K=0..7,16..23; 16-31 K=8..15,24..31).
//   C/D 16x16 f32: VGPR r = row r (lanes 0-15) and row 8+r (lanes 16-31).
// ---------------------------------------------------------------------------
__global__ void __launch_bounds__(256)
surv_pair_kernel(const float* __restrict__ t,
                 const _Float16* __restrict__ whi,
                 const _Float16* __restrict__ wlo,
                 float* __restrict__ pairSum,
                 int B) {
    __shared__ float    sh_t[PANEL];
    __shared__ _Float16 sh_hi[PANEL];
    __shared__ _Float16 sh_lo[PANEL];

    const int tid  = threadIdx.x;
    const int lane = tid & 31;
    const int wave = tid >> 5;
    const int hiL  = (lane >> 4) & 1;      // lane-group select
    const int m    = lane & 15;            // A-matrix row within the wave tile
    const int iRow = blockIdx.x * 128 + wave * 16 + m;
    const float ti = t[iRow];

    v8f acc = {};                          // f32 16x16 accumulator (8 VGPRs)

    for (int p0 = 0; p0 < B; p0 += PANEL) {
        __syncthreads();
        for (int k = tid; k < PANEL; k += 256) {
            sh_t[k]  = t[p0 + k];
            sh_hi[k] = whi[p0 + k];
            sh_lo[k] = wlo[p0 + k];
        }
        if (p0 + PANEL < B) {
            __builtin_prefetch(&t[p0 + PANEL + tid * 4], 0, 0);  // global_prefetch_b8
        }
        __syncthreads();

        for (int j = 0; j < PANEL; j += 32) {
            v16h a, bh, bl;
#pragma unroll
            for (int v = 0; v < 8; ++v) {
                // A-matrix K indices for this lane/VGPR (ISA 16-bit A 16x32 layout)
                const int ka = j + ((v < 4) ? 0 : 16) + (hiL ? 8 : 0) + 2 * (v & 3);
                a[2 * v]     = (_Float16)((sh_t[ka]     > ti) ? 1.0f : 0.0f);
                a[2 * v + 1] = (_Float16)((sh_t[ka + 1] > ti) ? 1.0f : 0.0f);
                // B-matrix: broadcast w across columns; same K order as A (lanes16-31 +8 / +24)
                const int kb = j + ((v < 4) ? 0 : 16) + (hiL ? 8 : 0) + 2 * (v & 3);
                bh[2 * v]     = sh_hi[kb];
                bh[2 * v + 1] = sh_hi[kb + 1];
                bl[2 * v]     = sh_lo[kb];
                bl[2 * v + 1] = sh_lo[kb + 1];
            }
            // D = A*B + C  (8-arg probe-confirmed signature)
            acc = __builtin_amdgcn_wmma_f32_16x16x32_f16(
                false, a, false, bh, (short)0, acc, false, false);
            acc = __builtin_amdgcn_wmma_f32_16x16x32_f16(
                false, a, false, bl, (short)0, acc, false, false);
        }
    }

    // All 16 D columns are identical (B was a broadcast); column 0 lives on
    // lane 0 (rows 0..7 in acc[0..7]) and lane 16 (rows 8..15).
    const int iBase = blockIdx.x * 128 + wave * 16;
    if (lane == 0) {
#pragma unroll
        for (int r = 0; r < 8; ++r) pairSum[iBase + r] = acc[r];
    } else if (lane == 16) {
#pragma unroll
        for (int r = 0; r < 8; ++r) pairSum[iBase + 8 + r] = acc[r];
    }
}

// ---------------------------------------------------------------------------
// Kernel 3: deterministic final reduction (fixed strided order + fixed tree).
//   valid_i = (c_i==0) && (pairSum_i > 0)   [sum>0 <=> any(pair_mask) exactly,
//                                            since every w_j > e^-4 > 0]
//   contrib_i = log(pairSum_i) - risk_i
//   out = mean(nll) + 0.5 * (count>0 ? sum(contrib)/max(count,1) : 0)
// ---------------------------------------------------------------------------
__global__ void surv_reduce_kernel(const float* __restrict__ risk,
                                   const float* __restrict__ nll,
                                   const int* __restrict__ c,
                                   const float* __restrict__ pairSum,
                                   float* __restrict__ out,
                                   int B) {
    __shared__ float sN[256];
    __shared__ float sC[256];
    __shared__ int   sK[256];
    const int tid = threadIdx.x;

    float nSum = 0.0f, cSum = 0.0f;
    int   k = 0;
    for (int i = tid; i < B; i += 256) {
        nSum += nll[i];
        float s = pairSum[i];
        if (c[i] == 0 && s > 0.0f) {
            cSum += __logf(s) - risk[i];
            ++k;
        }
    }
    sN[tid] = nSum; sC[tid] = cSum; sK[tid] = k;
    __syncthreads();
    for (int off = 128; off > 0; off >>= 1) {
        if (tid < off) {
            sN[tid] += sN[tid + off];
            sC[tid] += sC[tid + off];
            sK[tid] += sK[tid + off];
        }
        __syncthreads();
    }
    if (tid == 0) {
        float cnt  = (float)sK[0];
        float rank = (cnt > 0.0f) ? (sC[0] / fmaxf(cnt, 1.0f)) : 0.0f;
        out[0] = sN[0] / (float)B + 0.5f * rank;
    }
}

// ---------------------------------------------------------------------------
extern "C" void kernel_launch(void* const* d_in, const int* in_sizes, int n_in,
                              void* d_out, int out_size, void* d_ws, size_t ws_size,
                              hipStream_t stream) {
    const float* outputs = (const float*)d_in[0];  // [B, 4]
    const float* t       = (const float*)d_in[1];  // [B]
    const int*   y       = (const int*)d_in[2];    // [B]
    const int*   c       = (const int*)d_in[3];    // [B]
    const int B = in_sizes[1];                     // 8192 (multiple of PANEL & 128)

    // Workspace layout (floats): risk | nll | pairSum | whi(f16) | wlo(f16)
    float*    ws      = (float*)d_ws;
    float*    risk    = ws;
    float*    nll     = ws + B;
    float*    pairSum = ws + 2 * B;
    _Float16* whi     = (_Float16*)(ws + 3 * B);
    _Float16* wlo     = whi + B;

    surv_row_kernel<<<(B + 255) / 256, 256, 0, stream>>>(
        outputs, y, c, risk, whi, wlo, nll, B);

    surv_pair_kernel<<<B / 128, 256, 0, stream>>>(
        t, whi, wlo, pairSum, B);

    surv_reduce_kernel<<<1, 256, 0, stream>>>(
        risk, nll, c, pairSum, (float*)d_out, B);
}
